// RecoveryDecoder_73237782331556
// MI455X (gfx1250) — matrix operand order repaired and checked by reading, not compiled, not run on hardware
//
#include <hip/hip_runtime.h>
#include <hip/hip_bf16.h>
#include <cstdint>

// ---------------------------------------------------------------------------
// Types for CDNA5 WMMA (wave32): bf16 A/B (16 x bf16 = 8 VGPRs), f32 C/D (8 VGPRs)
// ---------------------------------------------------------------------------
typedef __attribute__((ext_vector_type(16))) __bf16 v16bf;
typedef __attribute__((ext_vector_type(8)))  float  v8f;
typedef __attribute__((ext_vector_type(4))) unsigned int v4u;
typedef __attribute__((ext_vector_type(8))) int v8i;
typedef __attribute__((ext_vector_type(4))) int v4i;

#if defined(__has_builtin)
#if __has_builtin(__builtin_amdgcn_tensor_load_to_lds) && __has_builtin(__builtin_amdgcn_s_wait_tensorcnt)
#define HAS_TDM 1
#endif
#endif
#ifndef HAS_TDM
#define HAS_TDM 0
#endif

union FragBF {
    v16bf v;
    unsigned short s[16];
    uint4 q[2];
};

union Chunk8 {
    uint4 q;
    unsigned short s[8];
};

__device__ __forceinline__ unsigned short f2bf(float f) {
    unsigned int u = __float_as_uint(f);
    u += 0x7FFFu + ((u >> 16) & 1u);   // round-to-nearest-even
    return (unsigned short)(u >> 16);
}

#if HAS_TDM
// Tensor Data Mover: 2D tile (tile_w x tile_h elements, 2-byte data) from global
// (row stride `stride` elements) into LDS at `lds_addr`, padding 16B after every
// 128B row so LDS row stride = tile_w + 8 shorts (16B-aligned, conflict-spread).
// D# layout per CDNA5 ISA ch.8 (group0: count/lds/global/type, group1: dims).
__device__ __forceinline__ void tdm_load_2d(unsigned lds_addr, unsigned long long ga,
                                            unsigned tile_w, unsigned tile_h,
                                            unsigned stride, unsigned dim0, unsigned dim1) {
    v4u g0 = { 1u,                                        // count=1, user desc
               lds_addr,                                  // lds_addr
               (unsigned)(ga & 0xFFFFFFFFull),            // global_addr lo
               (unsigned)((ga >> 32) & 0x01FFFFFFull) | 0x80000000u };  // hi | type=2
    v8i g1 = { (int)(0x00010000u            // data_size = 1 (2 bytes)
                   | 0x00100000u            // pad_enable
                   | (4u << 22)             // pad_interval: 32 dwords (128B)
                   | (3u << 25)),           // pad_amount: 4 dwords (16B)
               (int)((dim0 & 0xFFFFu) << 16),                             // tensor_dim0 lo
               (int)(((dim0 >> 16) & 0xFFFFu) | ((dim1 & 0xFFFFu) << 16)),// dim0 hi | dim1 lo
               (int)(((dim1 >> 16) & 0xFFFFu) | ((tile_w & 0xFFFFu) << 16)), // dim1 hi | tile_dim0
               (int)(tile_h & 0xFFFFu),                                   // tile_dim1 (tile_dim2=0)
               (int)stride,                                               // tensor_dim0_stride lo
               0, 0 };
    v4i gz = { 0, 0, 0, 0 };
#if __clang_major__ >= 23
    v8i gz8 = { 0, 0, 0, 0, 0, 0, 0, 0 };
    __builtin_amdgcn_tensor_load_to_lds(g0, g1, gz, gz, gz8, 0);
#else
    __builtin_amdgcn_tensor_load_to_lds(g0, g1, gz, gz, 0);
#endif
}
#endif

// ---------------------------------------------------------------------------
// Elementwise f32 -> bf16 cast
// ---------------------------------------------------------------------------
__global__ __launch_bounds__(256)
void cast_bf16_kernel(const float* __restrict__ in, unsigned short* __restrict__ out,
                      long long n) {
    long long base = (long long)blockIdx.x * 1024 + threadIdx.x;
#pragma unroll
    for (int j = 0; j < 4; ++j) {
        long long idx = base + (long long)j * 256;
        if (idx < n) out[idx] = f2bf(in[idx]);
    }
}

// ---------------------------------------------------------------------------
// Tiled GEMM: C[M,N] = A[M,K](bf16) @ B[K,N](bf16) + bias, optional ReLU.
// Block = 256 threads (8 waves), tile 128(m) x 256(n), K-depth 64.
// Waves 2(m) x 4(n); each wave owns 64x64 = 4x4 WMMA tiles (128 acc VGPRs).
// A tile staged by the Tensor Data Mover (padded rows); B tile staged
// *transposed* so B fragments are contiguous ds_load_b128 pairs.
// ---------------------------------------------------------------------------
__global__ __launch_bounds__(256)
void gemm_bf16_kernel(const unsigned short* __restrict__ A,
                      const unsigned short* __restrict__ Bw,
                      const float* __restrict__ bias,
                      float* __restrict__ Cf,
                      unsigned short* __restrict__ Cbf,
                      int M, int N, int K, int relu) {
    __shared__ unsigned short sA[128][72];    // row-major, row stride 144B
    __shared__ unsigned short sBt[256][72];   // B transposed: [n][k], stride 144B

    const int tid   = threadIdx.x;
    const int lane  = tid & 31;
    const int wave  = tid >> 5;
    const int wm    = wave & 1;          // 64-row half
    const int wn    = wave >> 1;         // 64-col quarter
    const int lh    = lane & 15;
    const int khalf = (lane >> 4) * 8;   // A-operand K sub-offset
    const int kbase = (lane >> 4) * 16;  // unused (B fragments also use khalf)
    const int m0    = blockIdx.y * 128;
    const int n0    = blockIdx.x * 256;

    v8f acc[4][4];
#pragma unroll
    for (int im = 0; im < 4; ++im)
#pragma unroll
        for (int jn = 0; jn < 4; ++jn)
#pragma unroll
            for (int e = 0; e < 8; ++e) acc[im][jn][e] = 0.0f;

    for (int k0 = 0; k0 < K; k0 += 64) {
        __syncthreads();
#if HAS_TDM
        if (wave == 0) {
            unsigned long long ga = (unsigned long long)(uintptr_t)A
                + ((unsigned long long)(unsigned)m0 * (unsigned)K + (unsigned)k0) * 2ull;
            tdm_load_2d((unsigned)(uintptr_t)&sA[0][0], ga,
                        64u, 128u, (unsigned)K, (unsigned)K, (unsigned)M);
            __builtin_amdgcn_s_wait_tensorcnt(0);
        }
#else
        // manual A staging: 128 x 64 bf16 = 1024 x 16B chunks / 256 threads
#pragma unroll
        for (int i = 0; i < 4; ++i) {
            int c   = tid + i * 256;
            int row = c >> 3;
            int off = (c & 7) * 8;
            *(uint4*)&sA[row][off] =
                *(const uint4*)&A[(size_t)(m0 + row) * K + k0 + off];
        }
#endif
        // B staging, transposed: 64(k) x 256(n) = 2048 x 16B chunks / 256 threads
#pragma unroll
        for (int i = 0; i < 8; ++i) {
            int c    = tid + i * 256;
            int krow = c >> 5;
            int noff = (c & 31) * 8;
            Chunk8 u;
            u.q = *(const uint4*)&Bw[(size_t)(k0 + krow) * N + n0 + noff];
#pragma unroll
            for (int j = 0; j < 8; ++j)
                sBt[noff + j][krow] = u.s[j];
        }
        if (k0 + 64 < K) {
            __builtin_prefetch(&A[(size_t)(m0 + (tid >> 1)) * K + k0 + 64], 0, 1);
            __builtin_prefetch(&Bw[(size_t)(k0 + 64 + (tid >> 3)) * N + n0], 0, 1);
        }
        __syncthreads();

#pragma unroll
        for (int ks = 0; ks < 64; ks += 32) {
            FragBF afr[4];
#pragma unroll
            for (int im = 0; im < 4; ++im) {
                int row = wm * 64 + im * 16 + lh;
                afr[im].q[0] = *(const uint4*)&sA[row][ks + khalf];
                afr[im].q[1] = *(const uint4*)&sA[row][ks + khalf + 16];
            }
#pragma unroll
            for (int jn = 0; jn < 4; ++jn) {
                FragBF bfr;
                int col = wn * 64 + jn * 16 + lh;
                bfr.q[0] = *(const uint4*)&sBt[col][ks + khalf];
                bfr.q[1] = *(const uint4*)&sBt[col][ks + khalf + 16];
#pragma unroll
                for (int im = 0; im < 4; ++im)
                    acc[im][jn] = __builtin_amdgcn_wmma_f32_16x16x32_bf16(
                        false, afr[im].v, false, bfr.v,
                        (short)0, acc[im][jn], false, false);
            }
        }
    }

    // ---- epilogue: bias (+ReLU), write f32 and/or bf16
    const int hi8 = (lane >> 4) * 8;
#pragma unroll
    for (int im = 0; im < 4; ++im) {
#pragma unroll
        for (int jn = 0; jn < 4; ++jn) {
            int col = n0 + wn * 64 + jn * 16 + lh;
            float bv = bias ? bias[col] : 0.0f;
#pragma unroll
            for (int e = 0; e < 8; ++e) {
                int row = m0 + wm * 64 + im * 16 + hi8 + e;
                float v = acc[im][jn][e] + bv;
                if (relu) v = fmaxf(v, 0.0f);
                size_t idx = (size_t)row * N + col;
                if (Cf)  Cf[idx]  = v;
                if (Cbf) Cbf[idx] = f2bf(v);
            }
        }
    }
}

// ---------------------------------------------------------------------------
// Fused flash attention (bf16 in, bf16 out), hd = 64, C = NH*hd = 1024.
// Block = 128 threads (4 waves); block owns a 64-row q tile, wave owns 16 rows.
// K staged by TDM (row-major, padded); V staged transposed so PV B-fragments
// are contiguous ds_load_b128 pairs. Online softmax; S and PV on WMMA.
// grid = (Nq/64, NH, B)
// ---------------------------------------------------------------------------
__global__ __launch_bounds__(128)
void attn_kernel(const unsigned short* __restrict__ Qb,
                 const unsigned short* __restrict__ Kb,
                 const unsigned short* __restrict__ Vb,
                 unsigned short* __restrict__ Obf,
                 int Nq, int Nk, float scale) {
    __shared__ unsigned short sK[64][72];    // [k][d]
    __shared__ unsigned short sVt[64][72];   // V transposed: [d][k]
    __shared__ unsigned short sP[64][72];    // [q][k]

    const int tid   = threadIdx.x;
    const int lane  = tid & 31;
    const int wave  = tid >> 5;
    const int lh    = lane & 15;
    const int khalf = (lane >> 4) * 8;
    const int kbase = (lane >> 4) * 16;
    const int hi8   = khalf;
    const int b  = blockIdx.z;
    const int h  = blockIdx.y;
    const int q0 = blockIdx.x * 64;

    FragBF aq[2];
    {
        int qrow = q0 + wave * 16 + lh;
        const unsigned short* qp = Qb + (size_t)(b * Nq + qrow) * 1024 + h * 64;
        aq[0].q[0] = *(const uint4*)&qp[khalf];
        aq[0].q[1] = *(const uint4*)&qp[khalf + 16];
        aq[1].q[0] = *(const uint4*)&qp[32 + khalf];
        aq[1].q[1] = *(const uint4*)&qp[32 + khalf + 16];
    }

    float mrow[8], lrow[8];
    v8f oacc[4];
#pragma unroll
    for (int e = 0; e < 8; ++e) { mrow[e] = -1e30f; lrow[e] = 0.0f; }
#pragma unroll
    for (int jd = 0; jd < 4; ++jd)
#pragma unroll
        for (int e = 0; e < 8; ++e) oacc[jd][e] = 0.0f;

    for (int kb = 0; kb < Nk; kb += 64) {
        __syncthreads();
#if HAS_TDM
        if (wave == 0) {
            unsigned long long ga = (unsigned long long)(uintptr_t)Kb
                + ((unsigned long long)(unsigned)(b * Nk + kb) * 1024ull
                   + (unsigned)(h * 64)) * 2ull;
            tdm_load_2d((unsigned)(uintptr_t)&sK[0][0], ga,
                        64u, 64u, 1024u, 1024u, (unsigned)Nk);
            __builtin_amdgcn_s_wait_tensorcnt(0);
        }
#else
#pragma unroll
        for (int i = 0; i < 4; ++i) {
            int c   = tid + i * 128;
            int row = c >> 3;
            int off = (c & 7) * 8;
            *(uint4*)&sK[row][off] =
                *(const uint4*)&Kb[(size_t)(b * Nk + kb + row) * 1024 + h * 64 + off];
        }
#endif
        // V staged transposed: [d][k]
#pragma unroll
        for (int i = 0; i < 4; ++i) {
            int c   = tid + i * 128;
            int row = c >> 3;                 // k within block
            int off = (c & 7) * 8;            // d offset
            Chunk8 u;
            u.q = *(const uint4*)&Vb[(size_t)(b * Nk + kb + row) * 1024 + h * 64 + off];
#pragma unroll
            for (int j = 0; j < 8; ++j)
                sVt[off + j][row] = u.s[j];
        }
        __syncthreads();

        // S = Q @ K^T
        v8f sacc[4];
#pragma unroll
        for (int jn = 0; jn < 4; ++jn) {
#pragma unroll
            for (int e = 0; e < 8; ++e) sacc[jn][e] = 0.0f;
            int kcol = jn * 16 + lh;
#pragma unroll
            for (int ch = 0; ch < 2; ++ch) {
                FragBF bk;
                bk.q[0] = *(const uint4*)&sK[kcol][ch * 32 + khalf];
                bk.q[1] = *(const uint4*)&sK[kcol][ch * 32 + khalf + 16];
                sacc[jn] = __builtin_amdgcn_wmma_f32_16x16x32_bf16(
                    false, aq[ch].v, false, bk.v, (short)0, sacc[jn], false, false);
            }
        }

        // online softmax per row
#pragma unroll
        for (int e = 0; e < 8; ++e) {
            float rm = -1e30f;
#pragma unroll
            for (int jn = 0; jn < 4; ++jn) {
                sacc[jn][e] *= scale;
                rm = fmaxf(rm, sacc[jn][e]);
            }
            rm = fmaxf(rm, __shfl_xor(rm, 1));
            rm = fmaxf(rm, __shfl_xor(rm, 2));
            rm = fmaxf(rm, __shfl_xor(rm, 4));
            rm = fmaxf(rm, __shfl_xor(rm, 8));
            float mn    = fmaxf(mrow[e], rm);
            float alpha = __expf(mrow[e] - mn);
            mrow[e] = mn;
            float ps = 0.0f;
#pragma unroll
            for (int jn = 0; jn < 4; ++jn) {
                float p = __expf(sacc[jn][e] - mn);
                sacc[jn][e] = p;
                ps += p;
            }
            ps += __shfl_xor(ps, 1);
            ps += __shfl_xor(ps, 2);
            ps += __shfl_xor(ps, 4);
            ps += __shfl_xor(ps, 8);
            lrow[e] = lrow[e] * alpha + ps;
#pragma unroll
            for (int jd = 0; jd < 4; ++jd) oacc[jd][e] *= alpha;
        }

        // P -> LDS (bf16), then O += P @ V
#pragma unroll
        for (int jn = 0; jn < 4; ++jn)
#pragma unroll
            for (int e = 0; e < 8; ++e)
                sP[wave * 16 + hi8 + e][jn * 16 + lh] = f2bf(sacc[jn][e]);
        __syncthreads();

#pragma unroll
        for (int ch = 0; ch < 2; ++ch) {
            FragBF ap;
            int prow = wave * 16 + lh;
            ap.q[0] = *(const uint4*)&sP[prow][ch * 32 + khalf];
            ap.q[1] = *(const uint4*)&sP[prow][ch * 32 + khalf + 16];
#pragma unroll
            for (int jd = 0; jd < 4; ++jd) {
                FragBF bv;
                int col = jd * 16 + lh;
                bv.q[0] = *(const uint4*)&sVt[col][ch * 32 + khalf];
                bv.q[1] = *(const uint4*)&sVt[col][ch * 32 + khalf + 16];
                oacc[jd] = __builtin_amdgcn_wmma_f32_16x16x32_bf16(
                    false, ap.v, false, bv.v, (short)0, oacc[jd], false, false);
            }
        }
    }

#pragma unroll
    for (int jd = 0; jd < 4; ++jd) {
#pragma unroll
        for (int e = 0; e < 8; ++e) {
            float v = oacc[jd][e] / lrow[e];
            size_t idx = (size_t)(b * Nq + q0 + wave * 16 + hi8 + e) * 1024
                       + h * 64 + jd * 16 + lh;
            Obf[idx] = f2bf(v);
        }
    }
}

// ---------------------------------------------------------------------------
// Fused residual + LayerNorm: out = LN(x + y) * g + b  (DIM = 1024, 1 row/block)
// ---------------------------------------------------------------------------
__global__ __launch_bounds__(256)
void add_ln_kernel(const float* __restrict__ x, const float* __restrict__ y,
                   const float* __restrict__ g, const float* __restrict__ bb,
                   float* __restrict__ outf, unsigned short* __restrict__ outbf) {
    __shared__ float red[16];
    const int row = blockIdx.x;
    const int tid = threadIdx.x;
    const float* xr = x + (size_t)row * 1024;
    const float* yr = y + (size_t)row * 1024;

    float v[4];
    float s = 0.0f, ss = 0.0f;
#pragma unroll
    for (int i = 0; i < 4; ++i) {
        int c = tid + i * 256;
        v[i] = xr[c] + yr[c];
        s  += v[i];
        ss += v[i] * v[i];
    }
    for (int m = 1; m < 32; m <<= 1) {
        s  += __shfl_xor(s, m);
        ss += __shfl_xor(ss, m);
    }
    const int wave = tid >> 5, lane = tid & 31;
    if (lane == 0) { red[wave] = s; red[8 + wave] = ss; }
    __syncthreads();
    if (tid == 0) {
        float ts = 0.0f, tss = 0.0f;
        for (int w = 0; w < 8; ++w) { ts += red[w]; tss += red[8 + w]; }
        red[0] = ts; red[8] = tss;
    }
    __syncthreads();
    const float mean = red[0] * (1.0f / 1024.0f);
    const float var  = red[8] * (1.0f / 1024.0f) - mean * mean;
    const float rstd = rsqrtf(var + 1e-5f);
#pragma unroll
    for (int i = 0; i < 4; ++i) {
        int c = tid + i * 256;
        float o = (v[i] - mean) * rstd * g[c] + bb[c];
        size_t idx = (size_t)row * 1024 + c;
        outf[idx] = o;
        if (outbf) outbf[idx] = f2bf(o);
    }
}

// ---------------------------------------------------------------------------
// Host orchestration
// ---------------------------------------------------------------------------
extern "C" void kernel_launch(void* const* d_in, const int* in_sizes, int n_in,
                              void* d_out, int out_size, void* d_ws, size_t ws_size,
                              hipStream_t stream) {
    (void)in_sizes; (void)n_in; (void)out_size; (void)ws_size;

    const int DIM_ = 1024, NH_ = 16, FF_ = 4096, L_ = 3, B_ = 8, NQ_ = 512, NK_ = 1024;
    const int Mq = B_ * NQ_;   // 4096
    const int Mk = B_ * NK_;   // 8192

    const float* q_in    = (const float*)d_in[0];
    const float* lat_in  = (const float*)d_in[1];
    const float* w_q_in  = (const float*)d_in[2];
    const float* b_q_in  = (const float*)d_in[3];
    const float* w_k_in  = (const float*)d_in[4];
    const float* b_k_in  = (const float*)d_in[5];
    const float* w_q_out = (const float*)d_in[6];
    const float* b_q_out = (const float*)d_in[7];
    const float* w_k_out = (const float*)d_in[8];
    const float* b_k_out = (const float*)d_in[9];

    char* ws = (char*)d_ws;
    size_t off = 0;
    auto alloc = [&](size_t bytes) -> void* {
        off = (off + 255) & ~(size_t)255;
        void* p = ws + off;
        off += bytes;
        return p;
    };

    auto cast_to = [&](const float* src, unsigned short* dst, size_t n) {
        long long ln = (long long)n;
        int grid = (int)((ln + 1023) / 1024);
        cast_bf16_kernel<<<grid, 256, 0, stream>>>(src, dst, ln);
    };
    auto castw = [&](const float* src, size_t n) -> unsigned short* {
        unsigned short* dst = (unsigned short*)alloc(n * 2);
        cast_to(src, dst, n);
        return dst;
    };

    // bf16 weights (~154 MB -> resident in the 192 MB L2 after first touch)
    unsigned short* wqin_bf  = castw(w_q_in,  (size_t)DIM_ * DIM_);
    unsigned short* wkin_bf  = castw(w_k_in,  (size_t)DIM_ * DIM_);
    unsigned short* wqout_bf = castw(w_q_out, (size_t)DIM_ * 2 * DIM_);
    unsigned short* wkout_bf = castw(w_k_out, (size_t)DIM_ * DIM_);

    unsigned short* wbf[2][6];
    for (int s = 0; s < 2; ++s) {
        int base = 10 + s * 16;
        wbf[s][0] = castw((const float*)d_in[base + 0],  (size_t)L_ * DIM_ * DIM_);
        wbf[s][1] = castw((const float*)d_in[base + 2],  (size_t)L_ * DIM_ * DIM_);
        wbf[s][2] = castw((const float*)d_in[base + 4],  (size_t)L_ * DIM_ * DIM_);
        wbf[s][3] = castw((const float*)d_in[base + 6],  (size_t)L_ * DIM_ * DIM_);
        wbf[s][4] = castw((const float*)d_in[base + 8],  (size_t)L_ * DIM_ * FF_);
        wbf[s][5] = castw((const float*)d_in[base + 10], (size_t)L_ * FF_ * DIM_);
    }

    float* qA   = (float*)alloc((size_t)Mq * DIM_ * 4);
    float* qB   = (float*)alloc((size_t)Mq * DIM_ * 4);
    float* latA = (float*)alloc((size_t)Mk * DIM_ * 4);
    float* latB = (float*)alloc((size_t)Mk * DIM_ * 4);
    unsigned short* qA_bf   = (unsigned short*)alloc((size_t)Mq * DIM_ * 2);
    unsigned short* qB_bf   = (unsigned short*)alloc((size_t)Mq * DIM_ * 2);
    unsigned short* latA_bf = (unsigned short*)alloc((size_t)Mk * DIM_ * 2);
    unsigned short* latB_bf = (unsigned short*)alloc((size_t)Mk * DIM_ * 2);
    unsigned short* tmp_bf  = (unsigned short*)alloc((size_t)Mk * DIM_ * 2);
    unsigned short* pq_bf   = (unsigned short*)alloc((size_t)Mk * DIM_ * 2);
    unsigned short* pk_bf   = (unsigned short*)alloc((size_t)Mk * DIM_ * 2);
    unsigned short* pv_bf   = (unsigned short*)alloc((size_t)Mk * DIM_ * 2);
    unsigned short* o_bf    = (unsigned short*)alloc((size_t)Mk * DIM_ * 2);
    float* attn_f = (float*)alloc((size_t)Mk * DIM_ * 4);
    float* t_f    = (float*)alloc((size_t)Mk * DIM_ * 4);
    unsigned short* t_bf   = (unsigned short*)alloc((size_t)Mk * DIM_ * 2);
    unsigned short* ff1_bf = (unsigned short*)alloc((size_t)Mk * FF_ * 2);
    float* ff2_f  = (float*)alloc((size_t)Mk * DIM_ * 4);

    auto gemm = [&](const unsigned short* A, const unsigned short* Bw, const float* bias,
                    float* Cf, unsigned short* Cbf, int M, int N, int K, int relu) {
        dim3 grid(N / 256, M / 128);
        gemm_bf16_kernel<<<grid, 256, 0, stream>>>(A, Bw, bias, Cf, Cbf, M, N, K, relu);
    };

    // input projections
    cast_to(q_in, tmp_bf, (size_t)Mq * DIM_);
    gemm(tmp_bf, wqin_bf, b_q_in, qA, qA_bf, Mq, DIM_, DIM_, 0);
    cast_to(lat_in, tmp_bf, (size_t)Mk * DIM_);
    gemm(tmp_bf, wkin_bf, b_k_in, latA, latA_bf, Mk, DIM_, DIM_, 0);

    auto layer = [&](const float* Xf, const unsigned short* Xbf, int Mx, int Nqx,
                     const unsigned short* Ybf, int My, int Nky,
                     int s, int li,
                     float* Xout_f, unsigned short* Xout_bf) {
        int base = 10 + s * 16;
        const float* bq  = (const float*)d_in[base + 1]  + (size_t)li * DIM_;
        const float* bk  = (const float*)d_in[base + 3]  + (size_t)li * DIM_;
        const float* bv  = (const float*)d_in[base + 5]  + (size_t)li * DIM_;
        const float* bo  = (const float*)d_in[base + 7]  + (size_t)li * DIM_;
        const float* b1  = (const float*)d_in[base + 9]  + (size_t)li * FF_;
        const float* b2  = (const float*)d_in[base + 11] + (size_t)li * DIM_;
        const float* n2g = (const float*)d_in[base + 12] + (size_t)li * DIM_;
        const float* n2b = (const float*)d_in[base + 13] + (size_t)li * DIM_;
        const float* n3g = (const float*)d_in[base + 14] + (size_t)li * DIM_;
        const float* n3b = (const float*)d_in[base + 15] + (size_t)li * DIM_;
        const unsigned short* wq = wbf[s][0] + (size_t)li * DIM_ * DIM_;
        const unsigned short* wk = wbf[s][1] + (size_t)li * DIM_ * DIM_;
        const unsigned short* wv = wbf[s][2] + (size_t)li * DIM_ * DIM_;
        const unsigned short* wo = wbf[s][3] + (size_t)li * DIM_ * DIM_;
        const unsigned short* w1 = wbf[s][4] + (size_t)li * DIM_ * FF_;
        const unsigned short* w2 = wbf[s][5] + (size_t)li * FF_ * DIM_;

        gemm(Xbf, wq, bq, nullptr, pq_bf, Mx, DIM_, DIM_, 0);
        gemm(Ybf, wk, bk, nullptr, pk_bf, My, DIM_, DIM_, 0);
        gemm(Ybf, wv, bv, nullptr, pv_bf, My, DIM_, DIM_, 0);

        dim3 ag(Nqx / 64, NH_, B_);
        attn_kernel<<<ag, 128, 0, stream>>>(pq_bf, pk_bf, pv_bf, o_bf, Nqx, Nky, 0.125f);

        gemm(o_bf, wo, bo, attn_f, nullptr, Mx, DIM_, DIM_, 0);
        add_ln_kernel<<<Mx, 256, 0, stream>>>(Xf, attn_f, n2g, n2b, t_f, t_bf);
        gemm(t_bf, w1, b1, nullptr, ff1_bf, Mx, FF_, DIM_, 1);
        gemm(ff1_bf, w2, b2, ff2_f, nullptr, Mx, DIM_, FF_, 0);
        add_ln_kernel<<<Mx, 256, 0, stream>>>(t_f, ff2_f, n3g, n3b, Xout_f, Xout_bf);
        (void)My;
    };

    float *qc = qA, *qn = qB, *latc = latA, *latn = latB;
    unsigned short *qcb = qA_bf, *qnb = qB_bf, *latcb = latA_bf, *latnb = latB_bf;
    for (int li = 0; li < L_; ++li) {
        layer(qc,   qcb,   Mq, NQ_, latcb, Mk, NK_, 0, li, qn,   qnb);
        layer(latc, latcb, Mk, NK_, qcb,   Mq, NQ_, 1, li, latn, latnb);
        float* tf;
        unsigned short* tb;
        tf = qc;   qc   = qn;   qn   = tf;
        tb = qcb;  qcb  = qnb;  qnb  = tb;
        tf = latc; latc = latn; latn = tf;
        tb = latcb; latcb = latnb; latnb = tb;
    }

    float* outq   = (float*)d_out;
    float* outlat = outq + (size_t)Mq * 2 * DIM_;
    gemm(qcb,   wqout_bf, b_q_out, outq,   nullptr, Mq, 2 * DIM_, DIM_, 0);
    gemm(latcb, wkout_bf, b_k_out, outlat, nullptr, Mk, DIM_,     DIM_, 0);
}
